// ChannelGroupConvUneven_11888469475659
// MI455X (gfx1250) — matrix operand order repaired and checked by reading, not compiled, not run on hardware
//
#include <hip/hip_runtime.h>

// ---------------------------------------------------------------------------
// Grouped "uneven channel" 3x3 conv as implicit GEMM on CDNA5 WMMA.
//   x:      (8, 256, 128, 128) f32   NCHW
//   weight: (320, 256, 3, 3)   f32   OIHW  (group g = oc/64 masks cin < param[g])
//   bias:   (320,)             f32
//   out:    (8, 320, 126, 126) f32
// conv = sum over 9 taps (r,s) of GEMM: M = oc, K = cin, N = pixels along W.
// Block: 64 oc (one full group -> uniform threshold) x 128 px of one row,
// 128 threads = 4 waves. Each wave: 4M x 2N register tile (64 oc x 32 px)
// -> 12 ds_b128 fragment loads per 8 v_wmma (1.5 ds/wmma).
// Staging packs cin-quads: 4 coalesced b32 global loads -> 1 ds_store_b64.
// ---------------------------------------------------------------------------

typedef __attribute__((ext_vector_type(16))) _Float16 v16h;
typedef __attribute__((ext_vector_type(8)))  _Float16 v8h;
typedef __attribute__((ext_vector_type(4)))  _Float16 v4h;
typedef __attribute__((ext_vector_type(8)))  float    v8f;

#define CIN   256
#define NOC   320
#define HIN   128
#define WIN   128
#define HOUT  126
#define WOUT  126

#define KPAD   40     // padded half-stride: 80B rows -> aligned b64/b128, conflict-free
#define MTILE  64     // oc per block = one group (threshold uniform)
#define NTILE  128    // px per block (4 waves x 32)
#define XCOLS  130    // NTILE + 2 halo for s in {0,1,2}

__global__ __launch_bounds__(128)
void ChannelGroupConvUneven_wmma(const float* __restrict__ x,
                                 const float* __restrict__ weight,
                                 const float* __restrict__ bias,
                                 const int*   __restrict__ param,
                                 float*       __restrict__ out)
{
    __shared__ _Float16 sA[9][MTILE][KPAD];   // [tap][oc][cin]  f16
    __shared__ _Float16 sX[3][XCOLS][KPAD];   // [row][col][cin] f16 (transposed)

    const int tid    = threadIdx.x;
    const int wave   = tid >> 5;
    const int lane   = tid & 31;
    const int laneM  = lane & 15;
    const int laneHi = lane >> 4;

    const int octile = blockIdx.x;        // 0..4  (= group id)
    const int row0   = blockIdx.y;        // 0..125 output row
    const int b      = blockIdx.z;        // batch

    const int oc0    = octile * MTILE;
    const int thresh = param[octile];     // mask cin < thresh (uniform for block)
    const int c0     = thresh & ~31;      // aligned start; ragged part zeroed in LDS

    const int n0 = wave * 32;             // this wave's 32-px strip

    v8f acc[4][2];
    #pragma unroll
    for (int mi = 0; mi < 4; ++mi)
        #pragma unroll
        for (int ni = 0; ni < 2; ++ni)
            acc[mi][ni] = (v8f){};

    for (int cin0 = c0; cin0 < CIN; cin0 += 32) {
        // ---- stage masked weight tile: 64 oc x 32 cin x 9 taps, f32 -> f16.
        // One cin-quad per iteration: 4 global b32 loads -> 1 ds_store_b64.
        // idx order (m, q, t): lanes sweep contiguous (cin,tap) memory per oc.
        for (int idx = tid; idx < MTILE * 8 * 9; idx += 128) {
            const int t  = idx % 9;
            const int r2 = idx / 9;
            const int q  = r2 & 7;        // cin quad
            const int m  = r2 >> 3;       // oc within tile
            const int cbase = cin0 + 4 * q;
            v4h h;
            #pragma unroll
            for (int j = 0; j < 4; ++j) {
                const int cin = cbase + j;
                float wv = weight[((oc0 + m) * CIN + cin) * 9 + t];
                if (cin < thresh) wv = 0.0f;
                h[j] = (_Float16)wv;
            }
            *(v4h*)&sA[t][m][4 * q] = h;
        }
        // ---- stage x tile: 3 rows x 130 cols x 32 cin, transposed to [row][col][cin].
        // One cin-quad per iteration; lanes sweep col -> coalesced global b32 loads.
        for (int idx = tid; idx < 3 * 8 * XCOLS; idx += 128) {
            const int col = idx % XCOLS;
            const int r2  = idx / XCOLS;
            const int q   = r2 & 7;       // cin quad
            const int row = r2 >> 3;      // 0..2
            v4h h;
            #pragma unroll
            for (int j = 0; j < 4; ++j) {
                float xv = 0.0f;
                if (col < WIN) {
                    xv = x[((b * CIN + (cin0 + 4 * q + j)) * HIN + (row0 + row)) * WIN + col];
                }
                h[j] = (_Float16)xv;
            }
            *(v4h*)&sX[row][col][4 * q] = h;
        }
        __syncthreads();

        // ---- 9 taps, 8 WMMAs each (4 M-tiles x 2 N-tiles), K = 32 cin
        #pragma unroll
        for (int t = 0; t < 9; ++t) {
            const int r = t / 3;
            const int s = t % 3;

            // A fragments (ISA 7.12.2, 16-bit 16x32):
            // lanes 0-15: K {0..7, 16..23}; lanes 16-31: K {8..15, 24..31}; M = lane%16
            v16h a[4];
            #pragma unroll
            for (int mi = 0; mi < 4; ++mi) {
                const v8h lo = *(const v8h*)&sA[t][16 * mi + laneM][8 * laneHi];
                const v8h hi = *(const v8h*)&sA[t][16 * mi + laneM][16 + 8 * laneHi];
                #pragma unroll
                for (int i = 0; i < 8; ++i) { a[mi][i] = lo[i]; a[mi][i + 8] = hi[i]; }
            }
            // B fragments (32x16): N = lane%16 (pixel), K {8*laneHi + 0..7, 16 + 8*laneHi + 0..7}
            v16h bf[2];
            #pragma unroll
            for (int ni = 0; ni < 2; ++ni) {
                const int col = n0 + 16 * ni + laneM + s;
                const v8h lo = *(const v8h*)&sX[r][col][8 * laneHi];
                const v8h hi = *(const v8h*)&sX[r][col][16 + 8 * laneHi];
                #pragma unroll
                for (int i = 0; i < 8; ++i) { bf[ni][i] = lo[i]; bf[ni][i + 8] = hi[i]; }
            }

            #pragma unroll
            for (int mi = 0; mi < 4; ++mi)
                #pragma unroll
                for (int ni = 0; ni < 2; ++ni)
                    acc[mi][ni] = __builtin_amdgcn_wmma_f32_16x16x32_f16(
                        false, a[mi], false, bf[ni],
                        (short)0, acc[mi][ni], false, false);
        }
        __syncthreads();
    }

    // ---- epilogue: C/D layout (VGPR r -> M = r + 8*laneHi, N = lane%16) -> NCHW
    #pragma unroll
    for (int mi = 0; mi < 4; ++mi) {
        #pragma unroll
        for (int ni = 0; ni < 2; ++ni) {
            const int px = n0 + ni * 16 + laneM;
            if (px < WOUT) {
                #pragma unroll
                for (int r = 0; r < 8; ++r) {
                    const int oc = oc0 + mi * 16 + r + 8 * laneHi;
                    out[((b * NOC + oc) * HOUT + row0) * WOUT + px] = acc[mi][ni][r] + bias[oc];
                }
            }
        }
    }
}

extern "C" void kernel_launch(void* const* d_in, const int* in_sizes, int n_in,
                              void* d_out, int out_size, void* d_ws, size_t ws_size,
                              hipStream_t stream) {
    const float* x      = (const float*)d_in[0];
    const float* weight = (const float*)d_in[1];
    const float* bias   = (const float*)d_in[2];
    const int*   param  = (const int*)d_in[3];
    float*       out    = (float*)d_out;

    dim3 grid(NOC / MTILE, HOUT, 8);   // (5 groups, 126 rows, 8 batch)
    dim3 block(128);
    ChannelGroupConvUneven_wmma<<<grid, block, 0, stream>>>(x, weight, bias, param, out);
}
